// Encoder_47442208751751
// MI455X (gfx1250) — matrix operand order repaired and checked by reading, not compiled, not run on hardware
//
#include <hip/hip_runtime.h>

// MI455X / gfx1250, wave32. Matrix ops via v_wmma_f32_16x16x32_f16,
// weight streaming via global_load_async_to_lds_b128 + s_wait_asynccnt.
typedef __attribute__((ext_vector_type(16))) _Float16 v16h;
typedef __attribute__((ext_vector_type(8)))  _Float16 v8h;
typedef __attribute__((ext_vector_type(8)))  float    v8f;

constexpr int NB = 64;    // batch
constexpr int NS = 512;   // sequence length
constexpr int NE = 512;   // embedding dim
constexpr int NH = 1024;  // hidden dim
constexpr int HPAD = NH + 8;   // LDS h rows: 2064B = 516 dwords -> conflict-free
constexpr int SPAD = 128 + 8;  // staged strip rows: 272B = 68 dwords -> conflict-free

// ---------------------------------------------------------------------------
// CDNA5 async global->LDS copy (ASYNCcnt-tracked); per-lane 16B transfer.
__device__ __forceinline__ void async_load_b128(unsigned lds_off, const void* gaddr) {
    asm volatile("global_load_async_to_lds_b128 %0, %1, off"
                 :: "v"(lds_off), "v"(gaddr) : "memory");
}
#define WAIT_ASYNC_16() asm volatile("s_wait_asynccnt 0x10" ::: "memory")
#define WAIT_ASYNC_0()  asm volatile("s_wait_asynccnt 0x0"  ::: "memory")
#define WAIT_DS_0()     asm volatile("s_wait_dscnt 0x0"     ::: "memory")

// Stage one [32 x 128] f16 strip of W (rows k0..k0+31, cols n0w..n0w+127) into LDS.
// 16 async ops; op i moves rows 2i (lanes 0-15) and 2i+1 (lanes 16-31), 512B each.
__device__ __forceinline__ void issue_chunk(const _Float16* stagebuf,
                                            const _Float16* __restrict__ w,
                                            int k0, int lane, int n0w) {
    const int sub = lane >> 4;          // row parity handled by this lane
    const int cb  = (lane & 15) * 8;    // column (halves); 8 halves = 16B
    const unsigned lbase = (unsigned)(size_t)stagebuf;   // flat[31:0] == LDS offset
    #pragma unroll
    for (int i = 0; i < 16; ++i) {
        const int r = i * 2 + sub;
        unsigned loff = lbase + (unsigned)(r * SPAD + cb) * 2u;
        const void* g = (const void*)(w + (size_t)(k0 + r) * NH + n0w + cb);
        async_load_b128(loff, g);
    }
}

// acc[nt] += h[16 x 1024] @ W[:, n0w .. n0w+127], K-chunked, double-buffered LDS stage.
__device__ __forceinline__ void recur_gemm(v8f (&acc)[8],
                                           const _Float16 (&h)[16][HPAD],
                                           const _Float16* __restrict__ w,
                                           _Float16 (&stage)[2][32][SPAD],
                                           int lane, int n0w, int arow, int akof) {
    WAIT_DS_0();                                    // prior reads of stage[0] done
    issue_chunk(&stage[0][0][0], w, 0, lane, n0w);  // prologue: chunk 0 -> buf 0
    for (int kc = 0; kc < NH / 32; ++kc) {
        const int k0  = kc * 32;
        const int buf = kc & 1;
        if (kc + 1 < NH / 32) {
            WAIT_DS_0();                            // reads of stage[buf^1] retired
            issue_chunk(&stage[buf ^ 1][0][0], w, k0 + 32, lane, n0w);
            WAIT_ASYNC_16();                        // in-order: chunk kc has landed
        } else {
            WAIT_ASYNC_0();
        }
        union { v16h v; v8h h2[2]; } au;            // A frag from h state (16B-aligned rows)
        au.h2[0] = *(const v8h*)&h[arow][k0 + akof];
        au.h2[1] = *(const v8h*)&h[arow][k0 + akof + 8];
        #pragma unroll
        for (int nt = 0; nt < 8; ++nt) {            // B frag: lane n = col n, element s = k 16h+s
            union { v16h v; v8h h2[2]; } bu;
            #pragma unroll
            for (int s = 0; s < 16; ++s)
                bu.v[s] = stage[buf][akof + s][nt * 16 + arow];
            acc[nt] = __builtin_amdgcn_wmma_f32_16x16x32_f16(
                false, au.v, false, bu.v, (short)0, acc[nt], false, false);
        }
    }
}

// ---------------------------------------------------------------------------
__global__ void cvt_f32_f16(const float* __restrict__ s, _Float16* __restrict__ d, int n) {
    int i = blockIdx.x * 256 + threadIdx.x;
    if (i < n) d[i] = (_Float16)s[i];
}

// Embedding gather -> f16: xf16[b*NS+s][e] = (f16) emb[src[b*NS+s]][e]
__global__ void gather_embed(const int* __restrict__ src, const float* __restrict__ emb,
                             _Float16* __restrict__ xf16) {
    int bs = blockIdx.x;
    long long row = (long long)src[bs];
    const float* e = emb + (size_t)row * NE;
    _Float16* o = xf16 + (size_t)bs * NE;
    for (int i = threadIdx.x; i < NE; i += 256) o[i] = (_Float16)e[i];
}

// ---------------------------------------------------------------------------
// Bulk input projection (hoisted out of the scan; fully parallel):
// xproj[s][b][n] = sum_k xf16[b*NS+s][k]*Wih0[k][n] + bih0[n] + bhh0[n]
__global__ void __launch_bounds__(256, 1)
xproj_gemm(const _Float16* __restrict__ xf16,   // [NB*NS][NE] f16
           const _Float16* __restrict__ w,      // [NE][NH]    f16
           const float* __restrict__ bih, const float* __restrict__ bhh,
           float* __restrict__ xproj)           // [NS][NB][NH] f32
{
    const int m0   = blockIdx.x * 16;
    const int wave = threadIdx.x >> 5;
    const int lane = threadIdx.x & 31;
    const int n0w  = wave * 128;
    const int arow = m0 + (lane & 15);
    const int akof = (lane >> 4) * 16;

    v8f acc[8];
    #pragma unroll
    for (int nt = 0; nt < 8; ++nt)
        #pragma unroll
        for (int i = 0; i < 8; ++i) acc[nt][i] = 0.0f;

    for (int kc = 0; kc < NE / 32; ++kc) {
        const int k0 = kc * 32;
        v16h a = *(const v16h*)(xf16 + (size_t)arow * NE + k0 + akof);
        __builtin_prefetch((const void*)(w + (size_t)(k0 + 32 + lane) * NH + n0w), 0, 0);
        #pragma unroll
        for (int nt = 0; nt < 8; ++nt) {
            v16h b;
            #pragma unroll
            for (int s = 0; s < 16; ++s)
                b[s] = w[(size_t)(k0 + akof + s) * NH + n0w + nt * 16 + (lane & 15)];
            acc[nt] = __builtin_amdgcn_wmma_f32_16x16x32_f16(
                false, a, false, b, (short)0, acc[nt], false, false);
        }
    }
    const int colT  = lane & 15;
    const int rbase = (lane >> 4) * 8;
    #pragma unroll
    for (int nt = 0; nt < 8; ++nt) {
        int n = n0w + nt * 16 + colT;
        float bias = bih[n] + bhh[n];
        #pragma unroll
        for (int i = 0; i < 8; ++i) {
            int r = m0 + rbase + i;             // r = b*NS + s
            int bb = r >> 9, ss = r & (NS - 1);
            xproj[((size_t)ss * NB + bb) * NH + n] = acc[nt][i] + bias;
        }
    }
}

// ---------------------------------------------------------------------------
// Persistent recurrence: 4 WGs, each owns 16 batch rows for all 512 steps.
// h0/h1 in LDS (f16); weights double-buffer streamed L2 -> LDS via async DMA.
__global__ void __launch_bounds__(256, 1)
rnn_scan(const float* __restrict__ xproj,       // [NS][NB][NH]
         const _Float16* __restrict__ whh0,
         const _Float16* __restrict__ wih1,
         const _Float16* __restrict__ whh1,
         const float* __restrict__ bih1, const float* __restrict__ bhh1,
         float* __restrict__ out)               // outputs [NB][NS][NH] ++ hidden [2][NB][NH]
{
    __shared__ _Float16 h0s[16][HPAD];                 // 33 KB
    __shared__ _Float16 h1s[16][HPAD];                 // 33 KB
    __shared__ _Float16 wstage[8][2][32][SPAD];        // 139 KB: per-wave double buffer

    const int bg   = blockIdx.x;
    const int wave = threadIdx.x >> 5;
    const int lane = threadIdx.x & 31;
    const int n0w  = wave * 128;
    const int colT  = lane & 15;
    const int rbase = (lane >> 4) * 8;
    const int arow  = lane & 15;
    const int akof  = (lane >> 4) * 16;

    for (int i = threadIdx.x; i < 16 * NH; i += 256) {
        h0s[i >> 10][i & (NH - 1)] = (_Float16)0.0f;
        h1s[i >> 10][i & (NH - 1)] = (_Float16)0.0f;
    }
    __syncthreads();

    v8f acc[8];
    for (int t = 0; t < NS; ++t) {
        // ---- layer 0: h0 = tanh(xproj[t] + h0 @ Whh0) ----------------------
        #pragma unroll
        for (int nt = 0; nt < 8; ++nt) {
            int n = n0w + nt * 16 + colT;
            const float* xp = xproj + ((size_t)t * NB + bg * 16 + rbase) * NH + n;
            #pragma unroll
            for (int i = 0; i < 8; ++i) acc[nt][i] = xp[(size_t)i * NH];
        }
        recur_gemm(acc, h0s, whh0, wstage[wave], lane, n0w, arow, akof);
        __syncthreads();                        // all reads of old h0 done
        #pragma unroll
        for (int nt = 0; nt < 8; ++nt) {
            int n = n0w + nt * 16 + colT;
            #pragma unroll
            for (int i = 0; i < 8; ++i)
                h0s[rbase + i][n] = (_Float16)tanhf(acc[nt][i]);
        }
        __syncthreads();                        // new h0 visible

        // ---- layer 1: h1 = tanh(b_ih1 + b_hh1 + h0 @ Wih1 + h1 @ Whh1) -----
        #pragma unroll
        for (int nt = 0; nt < 8; ++nt) {
            int n = n0w + nt * 16 + colT;
            float bias = bih1[n] + bhh1[n];
            #pragma unroll
            for (int i = 0; i < 8; ++i) acc[nt][i] = bias;
        }
        recur_gemm(acc, h0s, wih1, wstage[wave], lane, n0w, arow, akof);
        recur_gemm(acc, h1s, whh1, wstage[wave], lane, n0w, arow, akof);
        __syncthreads();                        // all reads of old h1 done
        #pragma unroll
        for (int nt = 0; nt < 8; ++nt) {
            int n = n0w + nt * 16 + colT;
            #pragma unroll
            for (int i = 0; i < 8; ++i) {
                float v = tanhf(acc[nt][i]);
                h1s[rbase + i][n] = (_Float16)v;
                int b = bg * 16 + rbase + i;
                out[((size_t)b * NS + t) * NH + n] = v;
            }
        }
        __syncthreads();                        // new h1 visible for next step
    }

    float* hid = out + (size_t)NB * NS * NH;
    for (int i = threadIdx.x; i < 16 * NH; i += 256) {
        int r = i >> 10, c = i & (NH - 1);
        int b = bg * 16 + r;
        hid[(size_t)b * NH + c]        = (float)h0s[r][c];
        hid[(size_t)(NB + b) * NH + c] = (float)h1s[r][c];
    }
}

// ---------------------------------------------------------------------------
extern "C" void kernel_launch(void* const* d_in, const int* in_sizes, int n_in,
                              void* d_out, int out_size, void* d_ws, size_t ws_size,
                              hipStream_t stream) {
    const int*   src  = (const int*)d_in[0];
    const float* emb  = (const float*)d_in[1];
    const float* Wih0 = (const float*)d_in[2];
    const float* Whh0 = (const float*)d_in[3];
    const float* bih0 = (const float*)d_in[4];
    const float* bhh0 = (const float*)d_in[5];
    const float* Wih1 = (const float*)d_in[6];
    const float* Whh1 = (const float*)d_in[7];
    const float* bih1 = (const float*)d_in[8];
    const float* bhh1 = (const float*)d_in[9];
    float* out = (float*)d_out;

    char* ws = (char*)d_ws;
    _Float16* xf16  = (_Float16*)ws;  ws += (size_t)NB * NS * NE * 2;   // 33.5 MB
    _Float16* wih0h = (_Float16*)ws;  ws += (size_t)NE * NH * 2;        //  1.0 MB
    _Float16* whh0h = (_Float16*)ws;  ws += (size_t)NH * NH * 2;        //  2.1 MB
    _Float16* wih1h = (_Float16*)ws;  ws += (size_t)NH * NH * 2;        //  2.1 MB
    _Float16* whh1h = (_Float16*)ws;  ws += (size_t)NH * NH * 2;        //  2.1 MB
    float*    xproj = (float*)ws;                                       // 134 MB

    cvt_f32_f16<<<(NE * NH + 255) / 256, 256, 0, stream>>>(Wih0, wih0h, NE * NH);
    cvt_f32_f16<<<(NH * NH + 255) / 256, 256, 0, stream>>>(Whh0, whh0h, NH * NH);
    cvt_f32_f16<<<(NH * NH + 255) / 256, 256, 0, stream>>>(Wih1, wih1h, NH * NH);
    cvt_f32_f16<<<(NH * NH + 255) / 256, 256, 0, stream>>>(Whh1, whh1h, NH * NH);

    gather_embed<<<NB * NS, 256, 0, stream>>>(src, emb, xf16);
    xproj_gemm<<<NB * NS / 16, 256, 0, stream>>>(xf16, wih0h, bih0, bhh0, xproj);
    rnn_scan<<<NB / 16, 256, 0, stream>>>(xproj, whh0h, wih1h, whh1h, bih1, bhh1, out);
}
